// DynamicSpatioTemporalMemory_61495341744493
// MI455X (gfx1250) — compile-verified, matrix operand order checked by reading, not run on hardware
//
#include <hip/hip_runtime.h>
#include <hip/hip_bf16.h>
#include <math.h>

// ---------------------------------------------------------------------------
// DynamicSpatioTemporalMemory for MI455X (gfx1250, wave32, WMMA)
// ---------------------------------------------------------------------------
#define Bn   32
#define Cn   256
#define HWn  4096        // 64*64
#define Dn   512
#define Dh   256         // D/2
#define Mn   256
#define NHn  8
#define HDn  64
#define EPSf 1e-5f

typedef _Float16 half8  __attribute__((ext_vector_type(8)));
typedef _Float16 half16 __attribute__((ext_vector_type(16)));
typedef float    float8 __attribute__((ext_vector_type(8)));

// ---------------------------------------------------------------------------
// Kernel 1: fold BN into conv weights (f32 -> f16), fold bias, zero pooled acc
// ---------------------------------------------------------------------------
__global__ __launch_bounds__(256) void k_prep(
    const float* __restrict__ w, const float* __restrict__ cb,
    const float* __restrict__ gamma, const float* __restrict__ beta,
    const float* __restrict__ mean, const float* __restrict__ var,
    _Float16* __restrict__ wf16, float* __restrict__ bias2,
    float* __restrict__ pooled)
{
    int idx = blockIdx.x * 256 + threadIdx.x;     // 0..65535
    int o   = idx >> 8;
    int c   = idx & 255;
    float scale = gamma[o] / sqrtf(var[o] + EPSf);
    wf16[idx] = (_Float16)(w[idx] * scale);
    if (c == 0) bias2[o] = (cb[o] - mean[o]) * scale + beta[o];
    if (idx < Bn * Dh) pooled[idx] = 0.0f;
}

// ---------------------------------------------------------------------------
// Kernel 2: fused 1x1-conv (GEMM, f16 WMMA) + BN(folded) + ReLU + global pool
//   grid = (B, HW/16), block = 256 (8 waves)
//   Per WG: one batch, one 16-wide spatial column tile, all 256 out channels.
// ---------------------------------------------------------------------------
__global__ __launch_bounds__(256) void k_encpool(
    const float* __restrict__ x, const _Float16* __restrict__ wf16,
    const float* __restrict__ bias2, float* __restrict__ pooled)
{
    __shared__ _Float16 xs[16][256];   // [col][k] f16 (B-fragment friendly)

    const int b    = blockIdx.x;
    const int n0   = blockIdx.y * 16;
    const int t    = threadIdx.x;
    const int lane = t & 31;
    const int wave = t >> 5;

    // --- stage x tile (C=256 x 16 cols) into LDS as f16, transposed ---
    {
        const float* xb = x + (size_t)b * Cn * HWn + n0;
        const int col = t & 15;
        const int cr  = t >> 4;           // 0..15
#pragma unroll
        for (int pass = 0; pass < 16; ++pass) {
            int c = pass * 16 + cr;
            xs[col][c] = (_Float16)xb[(size_t)c * HWn + col];
        }
    }
    __syncthreads();

    // wave owns row tiles [16*wave, +16) and [16*wave+128, +16)
    const int mb0 = wave * 16;
    const int mb1 = wave * 16 + 128;
    float8 acc0 = {};
    float8 acc1 = {};

    const int arow = lane & 15;              // A-matrix: M = lane%16
    const int koff = (lane < 16) ? 0 : 8;    // 16-bit frag K grouping
    const int bcol = lane & 15;              // B-matrix: N = lane%16

#pragma unroll
    for (int k0 = 0; k0 < 256; k0 += 32) {
        // B fragment from LDS  (halfs 0..7 = K[k0+koff..+7], 8..15 = +16)
        half16 bf;
        {
            half8 b0 = *(const half8*)(&xs[bcol][k0 + koff]);
            half8 b1 = *(const half8*)(&xs[bcol][k0 + koff + 16]);
#pragma unroll
            for (int i = 0; i < 8; ++i) { bf[i] = b0[i]; bf[i + 8] = b1[i]; }
        }
        // A fragment, tile 0 (weights hot in L2; prefetch next K chunk)
        {
            const _Float16* wr = wf16 + (size_t)(mb0 + arow) * 256 + k0 + koff;
            __builtin_prefetch(wr + 32, 0, 0);
            half8 a0 = *(const half8*)(wr);
            half8 a1 = *(const half8*)(wr + 16);
            half16 af;
#pragma unroll
            for (int i = 0; i < 8; ++i) { af[i] = a0[i]; af[i + 8] = a1[i]; }
            acc0 = __builtin_amdgcn_wmma_f32_16x16x32_f16(
                       false, af, false, bf, (short)0, acc0, false, false);
        }
        // A fragment, tile 1
        {
            const _Float16* wr = wf16 + (size_t)(mb1 + arow) * 256 + k0 + koff;
            __builtin_prefetch(wr + 32, 0, 0);
            half8 a0 = *(const half8*)(wr);
            half8 a1 = *(const half8*)(wr + 16);
            half16 af;
#pragma unroll
            for (int i = 0; i < 8; ++i) { af[i] = a0[i]; af[i + 8] = a1[i]; }
            acc1 = __builtin_amdgcn_wmma_f32_16x16x32_f16(
                       false, af, false, bf, (short)0, acc1, false, false);
        }
    }

    // C/D layout: lanes 0-15 -> M=r, lanes 16-31 -> M=r+8, N=lane&15.
    // bias + ReLU, reduce the 16 columns per row via shfl, atomically pool.
    const int radd = (lane >= 16) ? 8 : 0;
#pragma unroll
    for (int r = 0; r < 8; ++r) {
        int row0 = mb0 + r + radd;
        float v0 = fmaxf(acc0[r] + bias2[row0], 0.0f);
#pragma unroll
        for (int o = 1; o < 16; o <<= 1) v0 += __shfl_xor(v0, o, 16);
        if ((lane & 15) == 0) atomicAdd(&pooled[b * Dh + row0], v0);

        int row1 = mb1 + r + radd;
        float v1 = fmaxf(acc1[r] + bias2[row1], 0.0f);
#pragma unroll
        for (int o = 1; o < 16; o <<= 1) v1 += __shfl_xor(v1, o, 16);
        if ((lane & 15) == 0) atomicAdd(&pooled[b * Dh + row1], v1);
    }
}

// ---------------------------------------------------------------------------
// Kernel 3: pooled mean -> Linear(256->512) -> LayerNorm.  grid=B, block=512
// ---------------------------------------------------------------------------
__global__ __launch_bounds__(512) void k_enctail(
    const float* __restrict__ pooled, const float* __restrict__ lw,
    const float* __restrict__ lb, const float* __restrict__ lng,
    const float* __restrict__ lnb, float* __restrict__ enc)
{
    __shared__ float hv[Dh];
    __shared__ float red[Dn];
    __shared__ float red2[Dn];
    const int b = blockIdx.x, t = threadIdx.x;
    if (t < Dh) hv[t] = pooled[b * Dh + t] * (1.0f / (float)HWn);
    __syncthreads();
    const float* wr = lw + (size_t)t * Dh;
    float acc = lb[t];
    for (int c = 0; c < Dh; ++c) acc += hv[c] * wr[c];
    red[t] = acc; red2[t] = acc * acc;
    __syncthreads();
    for (int o = 256; o > 0; o >>= 1) {
        if (t < o) { red[t] += red[t + o]; red2[t] += red2[t + o]; }
        __syncthreads();
    }
    float mean = red[0] * (1.0f / (float)Dn);
    float var  = red2[0] * (1.0f / (float)Dn) - mean * mean;
    float rs   = 1.0f / sqrtf(var + EPSf);
    enc[b * Dn + t] = (acc - mean) * rs * lng[t] + lnb[t];
}

// ---------------------------------------------------------------------------
// Kernel 4: q/k/v projections. grid = 32 + 256 + 256 rows, block = 512
// ---------------------------------------------------------------------------
__global__ __launch_bounds__(512) void k_qkv(
    const float* __restrict__ enc, const float* __restrict__ mem,
    const float* __restrict__ aw, const float* __restrict__ ab,
    float* __restrict__ q, float* __restrict__ k, float* __restrict__ v)
{
    __shared__ float in[Dn];
    const int bid = blockIdx.x, t = threadIdx.x;
    const float* src; float* dst; int woff;
    if (bid < Bn)            { src = enc + bid * Dn;        dst = q + bid * Dn;        woff = 0;    }
    else if (bid < Bn + Mn)  { int m = bid - Bn;      src = mem + m * Dn; dst = k + m * Dn; woff = Dn;   }
    else                     { int m = bid - Bn - Mn; src = mem + m * Dn; dst = v + m * Dn; woff = 2*Dn; }
    in[t] = src[t];
    __syncthreads();
    const float* wr = aw + (size_t)(woff + t) * Dn;
    float acc = ab[woff + t];
    for (int j = 0; j < Dn; ++j) acc += in[j] * wr[j];
    dst[t] = acc;
}

// ---------------------------------------------------------------------------
// Kernel 5: attention (q_len=1) over memory bank. grid=B, block=256
// ---------------------------------------------------------------------------
__global__ __launch_bounds__(256) void k_attn(
    const float* __restrict__ q, const float* __restrict__ k,
    const float* __restrict__ v, float* __restrict__ ctx)
{
    __shared__ float qs[HDn];
    __shared__ float w[Mn];
    __shared__ float red[Mn];
    const int b = blockIdx.x, t = threadIdx.x;
    for (int h = 0; h < NHn; ++h) {
        if (t < HDn) qs[t] = q[b * Dn + h * HDn + t];
        __syncthreads();
        const float* kr = k + (size_t)t * Dn + h * HDn;
        float s = 0.0f;
        for (int d = 0; d < HDn; ++d) s += qs[d] * kr[d];
        s *= 0.125f;                         // 1/sqrt(64)
        red[t] = s;
        __syncthreads();
        for (int o = 128; o > 0; o >>= 1) {
            if (t < o) red[t] = fmaxf(red[t], red[t + o]);
            __syncthreads();
        }
        float mx = red[0];
        __syncthreads();
        float e = __expf(s - mx);
        w[t] = e; red[t] = e;
        __syncthreads();
        for (int o = 128; o > 0; o >>= 1) {
            if (t < o) red[t] += red[t + o];
            __syncthreads();
        }
        float inv = 1.0f / red[0];
        __syncthreads();
        const int d = t & 63, qa = t >> 6;
        float acc = 0.0f;
        for (int m = qa * 64; m < qa * 64 + 64; ++m)
            acc += w[m] * v[(size_t)m * Dn + h * HDn + d];
        red[t] = acc;
        __syncthreads();
        if (t < HDn)
            ctx[b * Dn + h * HDn + t] =
                (red[t] + red[t + 64] + red[t + 128] + red[t + 192]) * inv;
        __syncthreads();
    }
}

// ---------------------------------------------------------------------------
// Kernel 6: out-proj + gated fusion + channel-weight MLP. grid=B, block=512
// ---------------------------------------------------------------------------
__global__ __launch_bounds__(512) void k_fuse(
    const float* __restrict__ enc, const float* __restrict__ ctx,
    const float* __restrict__ ow, const float* __restrict__ ob,
    const float* __restrict__ gw, const float* __restrict__ gb,
    const float* __restrict__ r1w, const float* __restrict__ r1b,
    const float* __restrict__ r2w, const float* __restrict__ r2b,
    float* __restrict__ sw)
{
    __shared__ float cv[Dn];
    __shared__ float cat[2 * Dn];
    __shared__ float fu[Dn];
    __shared__ float r1[Dh];
    const int b = blockIdx.x, t = threadIdx.x;
    cv[t]  = ctx[b * Dn + t];
    cat[t] = enc[b * Dn + t];
    __syncthreads();
    const float* wr = ow + (size_t)t * Dn;
    float ret = ob[t];
    for (int j = 0; j < Dn; ++j) ret += cv[j] * wr[j];
    cat[Dn + t] = ret;
    __syncthreads();
    const float* gr = gw + (size_t)t * (2 * Dn);
    float g = gb[t];
    for (int j = 0; j < 2 * Dn; ++j) g += cat[j] * gr[j];
    g = 1.0f / (1.0f + __expf(-g));
    fu[t] = g * cat[t] + (1.0f - g) * cat[Dn + t];
    __syncthreads();
    if (t < Dh) {
        const float* rr = r1w + (size_t)t * Dn;
        float a = r1b[t];
        for (int j = 0; j < Dn; ++j) a += fu[j] * rr[j];
        r1[t] = fmaxf(a, 0.0f);
    }
    __syncthreads();
    if (t < Dh) {
        const float* rr = r2w + (size_t)t * Dh;
        float a = r2b[t];
        for (int j = 0; j < Dh; ++j) a += r1[j] * rr[j];
        sw[b * Cn + t] = 1.0f / (1.0f + __expf(-a));
    }
}

// ---------------------------------------------------------------------------
// Kernel 7: per-pixel channel avg & max.  grid=512, block=256
// ---------------------------------------------------------------------------
__global__ __launch_bounds__(256) void k_spstats(
    const float* __restrict__ x, float* __restrict__ sp)
{
    const int idx = blockIdx.x * 256 + threadIdx.x;   // 0..131071
    const int b = idx >> 12, hw = idx & 4095;
    const float* xb = x + (size_t)b * Cn * HWn + hw;
    float s = 0.0f, m = -3.4e38f;
    for (int c = 0; c < Cn; ++c) {
        float v = xb[(size_t)c * HWn];
        s += v; m = fmaxf(m, v);
    }
    sp[(size_t)b * 2 * HWn + hw]       = s * (1.0f / (float)Cn);
    sp[(size_t)b * 2 * HWn + HWn + hw] = m;
}

// ---------------------------------------------------------------------------
// Kernel 8: 7x7 conv (pad 3) on [avg,max] + sigmoid. grid=512, block=256
// ---------------------------------------------------------------------------
__global__ __launch_bounds__(256) void k_conv7(
    const float* __restrict__ sp, const float* __restrict__ w7,
    float* __restrict__ sa)
{
    __shared__ float wsm[98];
    const int t = threadIdx.x;
    if (t < 98) wsm[t] = w7[t];
    __syncthreads();
    const int idx = blockIdx.x * 256 + t;
    const int b = idx >> 12, hw = idx & 4095;
    const int hh = hw >> 6, wc = hw & 63;
    float acc = 0.0f;
    for (int i = 0; i < 2; ++i) {
        const float* pl = sp + (size_t)b * 2 * HWn + i * HWn;
        for (int kh = 0; kh < 7; ++kh) {
            int y = hh + kh - 3;
            if (y < 0 || y > 63) continue;
            for (int kw = 0; kw < 7; ++kw) {
                int xx = wc + kw - 3;
                if (xx < 0 || xx > 63) continue;
                acc += pl[y * 64 + xx] * wsm[i * 49 + kh * 7 + kw];
            }
        }
    }
    sa[idx] = 1.0f / (1.0f + __expf(-acc));
}

// ---------------------------------------------------------------------------
// Kernel 9: out = x * sw[b,c] * sa[b,hw].  grid-stride, bandwidth-bound.
// ---------------------------------------------------------------------------
__global__ __launch_bounds__(256) void k_final(
    const float* __restrict__ x, const float* __restrict__ sw,
    const float* __restrict__ sa, float* __restrict__ out)
{
    const size_t total = (size_t)Bn * Cn * HWn;
    for (size_t i = (size_t)blockIdx.x * 256 + threadIdx.x; i < total;
         i += (size_t)gridDim.x * 256) {
        size_t b   = i >> 20;            // Cn*HWn = 1<<20
        size_t rem = i & ((1u << 20) - 1);
        size_t c   = rem >> 12;
        size_t hw  = rem & 4095;
        out[i] = x[i] * sw[b * Cn + c] * sa[b * HWn + hw];
    }
}

// ---------------------------------------------------------------------------
extern "C" void kernel_launch(void* const* d_in, const int* in_sizes, int n_in,
                              void* d_out, int out_size, void* d_ws, size_t ws_size,
                              hipStream_t stream)
{
    const float* x        = (const float*)d_in[0];
    const float* mem      = (const float*)d_in[1];
    const float* conv_w   = (const float*)d_in[2];
    const float* conv_b   = (const float*)d_in[3];
    const float* bn_g     = (const float*)d_in[4];
    const float* bn_b     = (const float*)d_in[5];
    const float* bn_m     = (const float*)d_in[6];
    const float* bn_v     = (const float*)d_in[7];
    const float* lin_w    = (const float*)d_in[8];
    const float* lin_b    = (const float*)d_in[9];
    const float* ln_g     = (const float*)d_in[10];
    const float* ln_b     = (const float*)d_in[11];
    const float* ain_w    = (const float*)d_in[12];
    const float* ain_b    = (const float*)d_in[13];
    const float* aout_w   = (const float*)d_in[14];
    const float* aout_b   = (const float*)d_in[15];
    const float* gate_w   = (const float*)d_in[16];
    const float* gate_b   = (const float*)d_in[17];
    const float* rec_w1   = (const float*)d_in[18];
    const float* rec_b1   = (const float*)d_in[19];
    const float* rec_w2   = (const float*)d_in[20];
    const float* rec_b2   = (const float*)d_in[21];
    const float* spw      = (const float*)d_in[22];
    float* out            = (float*)d_out;

    char* ws = (char*)d_ws;
    _Float16* wf16 = (_Float16*)(ws);                 // 131072 B
    float* bias2   = (float*)(ws + 131072);           //   1024 B
    float* pooled  = (float*)(ws + 132096);           //  32768 B
    float* enc     = (float*)(ws + 164864);           //  65536 B
    float* q       = (float*)(ws + 230400);           //  65536 B
    float* kbuf    = (float*)(ws + 295936);           // 524288 B
    float* vbuf    = (float*)(ws + 820224);           // 524288 B
    float* ctx     = (float*)(ws + 1344512);          //  65536 B
    float* sw      = (float*)(ws + 1410048);          //  32768 B
    float* sp      = (float*)(ws + 1442816);          // 1048576 B
    float* sa      = (float*)(ws + 2491392);          // 524288 B

    k_prep<<<dim3(256), dim3(256), 0, stream>>>(conv_w, conv_b, bn_g, bn_b,
                                                bn_m, bn_v, wf16, bias2, pooled);
    k_encpool<<<dim3(Bn, HWn / 16), dim3(256), 0, stream>>>(x, wf16, bias2, pooled);
    k_enctail<<<dim3(Bn), dim3(512), 0, stream>>>(pooled, lin_w, lin_b, ln_g, ln_b, enc);
    k_qkv<<<dim3(Bn + 2 * Mn), dim3(512), 0, stream>>>(enc, mem, ain_w, ain_b, q, kbuf, vbuf);
    k_attn<<<dim3(Bn), dim3(256), 0, stream>>>(q, kbuf, vbuf, ctx);
    k_fuse<<<dim3(Bn), dim3(512), 0, stream>>>(enc, ctx, aout_w, aout_b, gate_w, gate_b,
                                               rec_w1, rec_b1, rec_w2, rec_b2, sw);
    k_spstats<<<dim3(512), dim3(256), 0, stream>>>(x, sp);
    k_conv7<<<dim3(512), dim3(256), 0, stream>>>(sp, spw, sa);
    k_final<<<dim3(16384), dim3(256), 0, stream>>>(x, sw, sa, out);
}
